// MaxEntIRLModel_86749749445358
// MI455X (gfx1250) — matrix operand order repaired and checked by reading, not compiled, not run on hardware
//
#include <hip/hip_runtime.h>

#define IN_DIM 5
#define HID    64
#define SEQ    30
#define NB     64          // sequences (rows) per workgroup
#define G4     256         // 4*HID gate columns
#define HS     72          // padded half-stride for h/feat buffers (bank-conflict-free)
#define GS     264         // padded half-stride for gate buffer
#define NTHREADS 256

typedef _Float16 v16h __attribute__((ext_vector_type(16)));
typedef _Float16 h8   __attribute__((ext_vector_type(8)));
typedef float    v8f  __attribute__((ext_vector_type(8)));

__device__ __forceinline__ float fast_sigmoid(float x) {
    return 1.0f / (1.0f + __expf(-x));
}
__device__ __forceinline__ float fast_tanh(float x) {
    return 2.0f / (1.0f + __expf(-2.0f * x)) - 1.0f;
}

// ---------------------------------------------------------------------------
// Prep: W_ih [256,64] -> f16 transposed [K=64][N=256]; same for W_hh; bias sum.
// ---------------------------------------------------------------------------
__global__ __launch_bounds__(256) void prep_kernel(
    const float* __restrict__ Wih, const float* __restrict__ Whh,
    const float* __restrict__ bih, const float* __restrict__ bhh,
    _Float16* __restrict__ WihT, _Float16* __restrict__ WhhT,
    float* __restrict__ biasSum)
{
    int i = blockIdx.x * 256 + threadIdx.x;
    if (i < HID * G4) {
        int k = i / G4;          // input-feature index 0..63
        int n = i % G4;          // gate column 0..255
        WihT[i] = (_Float16)Wih[n * HID + k];
        WhhT[i] = (_Float16)Whh[n * HID + k];
    }
    if (i < G4) biasSum[i] = bih[i] + bhh[i];
}

// ---------------------------------------------------------------------------
// Fused feature-net + LSTM + head. One WG = 64 sequences, all 30 steps.
// ---------------------------------------------------------------------------
__global__ __launch_bounds__(NTHREADS) void maxent_irl_kernel(
    const float* __restrict__ x,        // [N, SEQ, IN_DIM]
    const float* __restrict__ W1,       // [HID, IN_DIM]
    const float* __restrict__ b1,       // [HID]
    const _Float16* __restrict__ WihT,  // [64, 256] f16 (K-major)
    const _Float16* __restrict__ WhhT,  // [64, 256] f16 (K-major)
    const float* __restrict__ biasSum,  // [256] b_ih + b_hh
    const float* __restrict__ W2,       // [32, HID]
    const float* __restrict__ b2,       // [32]
    const float* __restrict__ W3,       // [32]
    const float* __restrict__ b3,       // [1]
    float* __restrict__ out)            // [N]
{
    __shared__ __align__(16) _Float16 gates[NB * GS];   // gate pre-activations
    __shared__ __align__(16) _Float16 hbuf[NB * HS];    // h_{t-1}, f16
    __shared__ __align__(16) _Float16 fbuf[NB * HS];    // feats_t, f16
    __shared__ __align__(16) float    w1l[HID * IN_DIM];
    __shared__ __align__(16) float    b1l[HID];
    __shared__ __align__(16) float    biasl[G4];        // reused for head partials

    const int tid  = threadIdx.x;
    const int wg   = blockIdx.x;
    const int lane = tid & 31;
    const int wv   = tid >> 5;      // wave 0..7
    const int ln16 = lane & 15;
    const int hi   = lane >> 4;     // 0 or 1

    // ---- stage small constants, zero h ----
    for (int i = tid; i < HID * IN_DIM; i += NTHREADS) w1l[i] = W1[i];
    if (tid < HID) b1l[tid] = b1[tid];
    biasl[tid] = biasSum[tid];
    for (int i = tid; i < NB * HS; i += NTHREADS) hbuf[i] = (_Float16)0.0f;

    // ---- preload this wave's B fragments (kept in VGPRs for all 30 steps) ----
    // B matrix is [K=64][N=256]; wave wv owns column tiles 2*wv and 2*wv+1.
    // B frags use contiguous k here (same as A): lane holds column n = l&15; lanes 16-31 take K+16.
    v16h BW[2][2][2];   // [ih/hh][coltile][kchunk]
    #pragma unroll
    for (int ct = 0; ct < 2; ++ct) {
        const int nbase = (2 * wv + ct) * 16 + ln16;
        #pragma unroll
        for (int kc = 0; kc < 2; ++kc) {
            v16h bi, bh;
            #pragma unroll
            for (int e = 0; e < 16; ++e) {
                const int k = kc * 32 + hi * 16 + e;
                bi[e] = WihT[k * G4 + nbase];
                bh[e] = WhhT[k * G4 + nbase];
            }
            BW[0][ct][kc] = bi;
            BW[1][ct][kc] = bh;
        }
    }

    // ---- elementwise-phase ownership: row er, 16 hidden units from ehu ----
    const int er  = tid >> 2;
    const int q   = tid & 3;
    const int ehu = q * 16;
    float c[16];
    float hreg[16];
    #pragma unroll
    for (int j = 0; j < 16; ++j) { c[j] = 0.0f; hreg[j] = 0.0f; }

    const float* xrow = x + (size_t)(wg * NB + er) * (SEQ * IN_DIM);

    __syncthreads();

    // ---- feats for t = 0 ----
    {
        float xv[IN_DIM];
        #pragma unroll
        for (int k = 0; k < IN_DIM; ++k) xv[k] = xrow[k];
        h8 fv0, fv1;
        #pragma unroll
        for (int j = 0; j < 16; ++j) {
            const int hu = ehu + j;
            float a = b1l[hu];
            #pragma unroll
            for (int k = 0; k < IN_DIM; ++k) a = fmaf(xv[k], w1l[hu * IN_DIM + k], a);
            const _Float16 f = (_Float16)fmaxf(a, 0.0f);
            if (j < 8) fv0[j] = f; else fv1[j - 8] = f;
        }
        *(h8*)&fbuf[er * HS + ehu]     = fv0;
        *(h8*)&fbuf[er * HS + ehu + 8] = fv1;
    }
    __syncthreads();

    // ================= time loop =================
    for (int t = 0; t < SEQ; ++t) {
        // ---- GEMM phase: gates = feats@W_ihT + h@W_hhT (WMMA f16->f32) ----
        #pragma unroll
        for (int rt = 0; rt < 4; ++rt) {
            const int row = rt * 16 + ln16;
            v16h Af[2], Ah[2];
            #pragma unroll
            for (int kc = 0; kc < 2; ++kc) {
                const int kb = kc * 32 + hi * 16;
                h8 flo = *(const h8*)&fbuf[row * HS + kb];
                h8 fhi = *(const h8*)&fbuf[row * HS + kb + 8];
                h8 hlo = *(const h8*)&hbuf[row * HS + kb];
                h8 hhi = *(const h8*)&hbuf[row * HS + kb + 8];
                v16h a, b;
                #pragma unroll
                for (int e = 0; e < 8; ++e) {
                    a[e] = flo[e]; a[e + 8] = fhi[e];
                    b[e] = hlo[e]; b[e + 8] = hhi[e];
                }
                Af[kc] = a; Ah[kc] = b;
            }
            // two independent accumulator chains (col tiles 2*wv, 2*wv+1)
            v8f acc0 = {}, acc1 = {};
            acc0 = __builtin_amdgcn_wmma_f32_16x16x32_f16(false, Af[0], false, BW[0][0][0], (short)0, acc0, false, false);
            acc1 = __builtin_amdgcn_wmma_f32_16x16x32_f16(false, Af[0], false, BW[0][1][0], (short)0, acc1, false, false);
            acc0 = __builtin_amdgcn_wmma_f32_16x16x32_f16(false, Af[1], false, BW[0][0][1], (short)0, acc0, false, false);
            acc1 = __builtin_amdgcn_wmma_f32_16x16x32_f16(false, Af[1], false, BW[0][1][1], (short)0, acc1, false, false);
            acc0 = __builtin_amdgcn_wmma_f32_16x16x32_f16(false, Ah[0], false, BW[1][0][0], (short)0, acc0, false, false);
            acc1 = __builtin_amdgcn_wmma_f32_16x16x32_f16(false, Ah[0], false, BW[1][1][0], (short)0, acc1, false, false);
            acc0 = __builtin_amdgcn_wmma_f32_16x16x32_f16(false, Ah[1], false, BW[1][0][1], (short)0, acc0, false, false);
            acc1 = __builtin_amdgcn_wmma_f32_16x16x32_f16(false, Ah[1], false, BW[1][1][1], (short)0, acc1, false, false);
            // C layout: lane l, vgpr r -> M = rt*16 + hi*8 + r, N = tile_n + (l&15)
            const int ncol0 = (2 * wv) * 16 + ln16;
            #pragma unroll
            for (int r2 = 0; r2 < 8; ++r2) {
                const int m = rt * 16 + hi * 8 + r2;
                gates[m * GS + ncol0]      = (_Float16)acc0[r2];
                gates[m * GS + ncol0 + 16] = (_Float16)acc1[r2];
            }
        }
        __syncthreads();

        // ---- elementwise phase: gate nonlinearities, c/h update ----
        {
            float z[4][16];
            #pragma unroll
            for (int g = 0; g < 4; ++g) {
                #pragma unroll
                for (int hb = 0; hb < 2; ++hb) {
                    h8 v = *(const h8*)&gates[er * GS + g * HID + ehu + hb * 8];
                    #pragma unroll
                    for (int e = 0; e < 8; ++e) z[g][hb * 8 + e] = (float)v[e];
                }
            }
            h8 hv0, hv1;
            #pragma unroll
            for (int j = 0; j < 16; ++j) {
                const int hu = ehu + j;
                const float vi = z[0][j] + biasl[hu];
                const float vf = z[1][j] + biasl[HID + hu];
                const float vg = z[2][j] + biasl[2 * HID + hu];
                const float vo = z[3][j] + biasl[3 * HID + hu];
                const float cn = fast_sigmoid(vf) * c[j] + fast_sigmoid(vi) * fast_tanh(vg);
                c[j] = cn;
                const float hn = fast_sigmoid(vo) * fast_tanh(cn);
                hreg[j] = hn;
                const _Float16 hf = (_Float16)hn;
                if (j < 8) hv0[j] = hf; else hv1[j - 8] = hf;
            }
            *(h8*)&hbuf[er * HS + ehu]     = hv0;
            *(h8*)&hbuf[er * HS + ehu + 8] = hv1;

            // feats for t+1
            if (t + 1 < SEQ) {
                if (t + 2 < SEQ) __builtin_prefetch(&xrow[(t + 2) * IN_DIM], 0, 1);
                float xv[IN_DIM];
                #pragma unroll
                for (int k = 0; k < IN_DIM; ++k) xv[k] = xrow[(t + 1) * IN_DIM + k];
                h8 fv0, fv1;
                #pragma unroll
                for (int j = 0; j < 16; ++j) {
                    const int hu = ehu + j;
                    float a = b1l[hu];
                    #pragma unroll
                    for (int k = 0; k < IN_DIM; ++k) a = fmaf(xv[k], w1l[hu * IN_DIM + k], a);
                    const _Float16 f = (_Float16)fmaxf(a, 0.0f);
                    if (j < 8) fv0[j] = f; else fv1[j - 8] = f;
                }
                *(h8*)&fbuf[er * HS + ehu]     = fv0;
                *(h8*)&fbuf[er * HS + ehu + 8] = fv1;
            }
        }
        __syncthreads();
    }

    // ---- stash final h (f32) into reused gate LDS: row stride 132 floats ----
    float* hfin = (float*)gates;
    #pragma unroll
    for (int j = 0; j < 16; ++j) hfin[er * (GS / 2) + ehu + j] = hreg[j];
    __syncthreads();

    // ---- head: Linear(64,32) -> ReLU -> Linear(32,1), 4 threads per row ----
    {
        const float* hr = hfin + er * (GS / 2);
        float part = 0.0f;
        #pragma unroll
        for (int jj = 0; jj < 8; ++jj) {
            const int j = q * 8 + jj;
            float a2 = b2[j];
            #pragma unroll
            for (int k2 = 0; k2 < HID; ++k2) a2 = fmaf(W2[j * HID + k2], hr[k2], a2);
            part = fmaf(W3[j], fmaxf(a2, 0.0f), part);
        }
        biasl[tid] = part;   // biasl is dead after the time loop; reuse for reduce
    }
    __syncthreads();
    if (tid < NB) {
        out[wg * NB + tid] = b3[0] + biasl[tid * 4] + biasl[tid * 4 + 1]
                                   + biasl[tid * 4 + 2] + biasl[tid * 4 + 3];
    }
}

// ---------------------------------------------------------------------------
extern "C" void kernel_launch(void* const* d_in, const int* in_sizes, int n_in,
                              void* d_out, int out_size, void* d_ws, size_t ws_size,
                              hipStream_t stream) {
    const float* x   = (const float*)d_in[0];
    const float* W1  = (const float*)d_in[1];
    const float* b1  = (const float*)d_in[2];
    const float* Wih = (const float*)d_in[3];
    const float* Whh = (const float*)d_in[4];
    const float* bih = (const float*)d_in[5];
    const float* bhh = (const float*)d_in[6];
    const float* W2  = (const float*)d_in[7];
    const float* b2  = (const float*)d_in[8];
    const float* W3  = (const float*)d_in[9];
    const float* b3  = (const float*)d_in[10];

    char* ws = (char*)d_ws;
    _Float16* WihT   = (_Float16*)(ws);
    _Float16* WhhT   = (_Float16*)(ws + 64 * 256 * sizeof(_Float16));
    float*    biasSm = (float*)(ws + 2 * 64 * 256 * sizeof(_Float16));

    prep_kernel<<<64, 256, 0, stream>>>(Wih, Whh, bih, bhh, WihT, WhhT, biasSm);

    const int nseq = 512 * 64;           // B * NTRAJ
    const int ngrid = nseq / NB;         // 512 workgroups
    maxent_irl_kernel<<<ngrid, NTHREADS, 0, stream>>>(
        x, W1, b1, WihT, WhhT, biasSm, W2, b2, W3, b3, (float*)d_out);
}
